// SophisticatedModel_11029476016752
// MI455X (gfx1250) — compile-verified, matrix operand order checked by reading, not run on hardware
//
#include <hip/hip_runtime.h>
#include <stdint.h>
#include <math.h>

// ---------------- problem constants (from reference) ----------------
#define NWN   20000
#define NEN   30000
#define ERN   200000
#define EEN   200000
#define HD    512
#define ECHD  4101
#define ECHP  4128      // ECH padded to multiple of 32 for WMMA K-loop
#define OUTD  250
#define NLAY  4
#define HH    (HD*HD)
#define SLOPE 0.2f

typedef __bf16 bf16;
typedef __bf16 v16bf __attribute__((ext_vector_type(16)));
typedef float  v8f   __attribute__((ext_vector_type(8)));

// Async global->LDS copy (CDNA5): use if this toolchain declares the builtins.
#if defined(__has_builtin)
#if __has_builtin(__builtin_amdgcn_global_load_async_to_lds_b128) && \
    __has_builtin(__builtin_amdgcn_s_wait_asynccnt)
#define HAS_ASYNC_LDS 1
#endif
#endif
#ifndef HAS_ASYNC_LDS
#define HAS_ASYNC_LDS 0
#endif

#if HAS_ASYNC_LDS
typedef int vint4_t __attribute__((vector_size(16)));
typedef __attribute__((address_space(1))) vint4_t* as1_v4i;
typedef __attribute__((address_space(3))) vint4_t* as3_v4i;
#endif

__device__ __forceinline__ bf16 f2bf(float f) {
  union { float f; uint32_t u; } a; a.f = f;
  uint32_t u = a.u;
  uint32_t r = u + 0x7FFFu + ((u >> 16) & 1u);   // round-to-nearest-even
  union { uint16_t s; bf16 b; } o; o.s = (uint16_t)(r >> 16);
  return o.b;
}

union Frag { v16bf v; uint4 q[2]; bf16 h[16]; };

// ---------------- WMMA GEMM: D = act((Cin + A*Bt^T + bias)*scale) ----------
// A  : [M, lda] f32 row-major (bf16-converted while staging to LDS), or
// Abf: [M, lda] bf16 row-major (used when non-null; direct 128b staging)
// Bt : [N, K] bf16 (pre-transposed weights, K multiple of 32, zero-padded)
// Cin may alias D (same geometry, ldd). All K multiples of 32; lda%4==0 (f32).
#define BM 128
#define BN 128
#define BK 32
#define LDT 40   // padded LDS row stride (bf16) -> conflict-free b128 frag loads

__device__ __forceinline__ v16bf frag_ld(const bf16* tile_base, int lane) {
  // 16-bit operand fragment layout (CDNA5 ISA 7.12.2):
  // lane<16 : row=lane,    K = {0..7} and {16..23}
  // lane>=16: row=lane-16, K = {8..15} and {24..31}
  int half = lane >> 4;
  int r    = lane & 15;
  const bf16* p = tile_base + r * LDT;
  Frag f;
  f.q[0] = *(const uint4*)(p + half * 8);
  f.q[1] = *(const uint4*)(p + 16 + half * 8);
  return f.v;
}

__global__ __launch_bounds__(256) void k_gemm(
    const float* A, const bf16* Abf, int lda,
    const bf16* __restrict__ Bt,
    const float* Cin, const float* __restrict__ bias,
    float* D, int ldd,
    int M, int N, int K, float scale, int leaky)
{
  __shared__ bf16 lds_a[2][BM * LDT];
  __shared__ bf16 lds_b[2][BN * LDT];

  int tid  = threadIdx.x;
  int lane = tid & 31;
  int wave = tid >> 5;          // 8 waves
  int wm   = wave >> 1;         // 0..3  -> 32 rows each
  int wn   = wave & 1;          // 0..1  -> 64 cols each
  int bm0  = blockIdx.y * BM;
  int bn0  = blockIdx.x * BN;

  int lrow = tid >> 1;          // 0..127
  int lcol = (tid & 1) * 16;    // 0 / 16

  v8f acc[2][4];
  #pragma unroll
  for (int i = 0; i < 2; i++)
    #pragma unroll
    for (int j = 0; j < 4; j++)
      #pragma unroll
      for (int e = 0; e < 8; e++) acc[i][j][e] = 0.0f;

  int nk = K / BK;

  auto stage = [&](int kt, int b) {
    int k0 = kt * BK;
    // ---- A tile ----
    {
      int gr = bm0 + lrow;
      Frag fa;
      if (Abf) {
        if (gr < M) {
          const uint4* p = (const uint4*)(Abf + (size_t)gr * lda + k0 + lcol);
          fa.q[0] = p[0]; fa.q[1] = p[1];
        } else {
          fa.q[0] = make_uint4(0,0,0,0); fa.q[1] = make_uint4(0,0,0,0);
        }
      } else if (gr < M) {
        const float4* p = (const float4*)(A + (size_t)gr * lda + k0 + lcol);
        float4 x0 = p[0], x1 = p[1], x2 = p[2], x3 = p[3];
        fa.h[0]=f2bf(x0.x); fa.h[1]=f2bf(x0.y); fa.h[2]=f2bf(x0.z); fa.h[3]=f2bf(x0.w);
        fa.h[4]=f2bf(x1.x); fa.h[5]=f2bf(x1.y); fa.h[6]=f2bf(x1.z); fa.h[7]=f2bf(x1.w);
        fa.h[8]=f2bf(x2.x); fa.h[9]=f2bf(x2.y); fa.h[10]=f2bf(x2.z); fa.h[11]=f2bf(x2.w);
        fa.h[12]=f2bf(x3.x); fa.h[13]=f2bf(x3.y); fa.h[14]=f2bf(x3.z); fa.h[15]=f2bf(x3.w);
      } else {
        #pragma unroll
        for (int j = 0; j < 16; j++) fa.h[j] = f2bf(0.0f);
      }
      *(uint4*)&lds_a[b][lrow * LDT + lcol]     = fa.q[0];
      *(uint4*)&lds_a[b][lrow * LDT + lcol + 8] = fa.q[1];
    }
    // ---- B^T tile (pure copy; async DMA to LDS when available) ----
    {
      int gn = bn0 + lrow;
      bf16* lp = &lds_b[b][lrow * LDT + lcol];
      if (gn < N) {
        const bf16* gp = Bt + (size_t)gn * K + k0 + lcol;
#if HAS_ASYNC_LDS
        __builtin_amdgcn_global_load_async_to_lds_b128(
            (as1_v4i)(__attribute__((address_space(1))) void*)(void*)gp,
            (as3_v4i)(__attribute__((address_space(3))) void*)(void*)lp, 0, 0);
        __builtin_amdgcn_global_load_async_to_lds_b128(
            (as1_v4i)(__attribute__((address_space(1))) void*)(void*)(gp + 8),
            (as3_v4i)(__attribute__((address_space(3))) void*)(void*)(lp + 8), 0, 0);
#else
        const uint4* p = (const uint4*)gp;
        ((uint4*)lp)[0] = p[0];
        *(uint4*)(lp + 8) = p[1];
#endif
      } else {
        ((uint4*)lp)[0] = make_uint4(0,0,0,0);
        *(uint4*)(lp + 8) = make_uint4(0,0,0,0);
      }
    }
  };

  stage(0, 0);

  for (int kt = 0; kt < nk; ++kt) {
    int cur = kt & 1;
#if HAS_ASYNC_LDS
    __builtin_amdgcn_s_wait_asynccnt(0);
#endif
    __syncthreads();
    if (kt + 1 < nk) stage(kt + 1, cur ^ 1);   // overlap next-tile loads with WMMAs

    v16bf af[2], bfv[4];
    #pragma unroll
    for (int i = 0; i < 2; i++) af[i]  = frag_ld(&lds_a[cur][(wm * 32 + i * 16) * LDT], lane);
    #pragma unroll
    for (int j = 0; j < 4; j++) bfv[j] = frag_ld(&lds_b[cur][(wn * 64 + j * 16) * LDT], lane);

    #pragma unroll
    for (int i = 0; i < 2; i++)
      #pragma unroll
      for (int j = 0; j < 4; j++)
        acc[i][j] = __builtin_amdgcn_wmma_f32_16x16x32_bf16(
            false, af[i], false, bfv[j], (short)0, acc[i][j], false, false);
  }

  // ---- epilogue ----
  int halfsel = lane >> 4;
  int ln = lane & 15;
  #pragma unroll
  for (int i = 0; i < 2; i++) {
    int rb = bm0 + wm * 32 + i * 16 + halfsel * 8;
    #pragma unroll
    for (int j = 0; j < 4; j++) {
      int c = bn0 + wn * 64 + j * 16 + ln;
      if (c < N) {
        #pragma unroll
        for (int v = 0; v < 8; v++) {
          int r = rb + v;
          if (r < M) {
            float val = acc[i][j][v];
            size_t off = (size_t)r * ldd + c;
            if (Cin)  val += Cin[off];
            if (bias) val += bias[c];
            val *= scale;
            if (leaky) val = val > 0.0f ? val : SLOPE * val;
            D[off] = val;
          }
        }
      }
    }
  }
}

// ---------------- weight / activation prep kernels ----------------
// out[n*Kpad + k] = bf16(W[k*N + n]), zero-pad k in [K, Kpad)
__global__ void k_wconvT(const float* __restrict__ W, bf16* __restrict__ out,
                         int K, int N, int Kpad) {
  long long total = (long long)N * Kpad;
  for (long long idx = blockIdx.x * 256LL + threadIdx.x; idx < total;
       idx += (long long)gridDim.x * 256) {
    int n = (int)(idx / Kpad), k = (int)(idx % Kpad);
    out[idx] = (k < K) ? f2bf(W[(size_t)k * N + n]) : f2bf(0.0f);
  }
}

// out[r*Kpad + k] = bf16(x[r*K + k]), zero-pad k in [K, Kpad)   (row-major copy)
__global__ void k_convpad(const float* __restrict__ x, bf16* __restrict__ out,
                          int M, int K, int Kpad) {
  long long total = (long long)M * Kpad;
  for (long long idx = blockIdx.x * 256LL + threadIdx.x; idx < total;
       idx += (long long)gridDim.x * 256) {
    int r = (int)(idx / Kpad), k = (int)(idx % Kpad);
    out[idx] = (k < K) ? f2bf(x[(size_t)r * K + k]) : f2bf(0.0f);
  }
}

// out[n*H + k] = bf16(sum_r W[r][k*H + n]), r = 0..2
__global__ void k_wsum3T(const float* __restrict__ W, bf16* __restrict__ out, int Hh) {
  int total = Hh * Hh;
  for (int idx = blockIdx.x * 256 + threadIdx.x; idx < total; idx += gridDim.x * 256) {
    int n = idx / Hh, k = idx % Hh;
    float s = W[(size_t)k * Hh + n] + W[(size_t)Hh * Hh + k * Hh + n]
            + W[2 * (size_t)Hh * Hh + k * Hh + n];
    out[idx] = f2bf(s);
  }
}

// o[i] = a[i] + a[n+i] + a[2n+i]
__global__ void k_add3(const float* __restrict__ a, float* __restrict__ o, int n) {
  for (int i = blockIdx.x * 256 + threadIdx.x; i < n; i += gridDim.x * 256)
    o[i] = a[i] + a[(size_t)n + i] + a[2 * (size_t)n + i];
}

__global__ void k_zero(float* p, long long n) {
  for (long long i = blockIdx.x * 256LL + threadIdx.x; i < n;
       i += (long long)gridDim.x * 256) p[i] = 0.0f;
}

// ---------------- graph scatter (segment-sum via L2 atomics) ----------------
__global__ __launch_bounds__(256) void k_scatter_win(
    const int* __restrict__ src, const int* __restrict__ dst, const int* __restrict__ ij,
    const float* __restrict__ hw, const float* __restrict__ he,
    float* agg, float* cnt, int ne) {
  int e = blockIdx.x;
  if (e >= ne) return;
  int s = src[e], d = dst[e], j = ij[e];
  const float* ps = hw + (size_t)s * HD;
  const float* pe = he + (size_t)j * HD;
  float* pd = agg + (size_t)d * HD;
  for (int c = threadIdx.x; c < HD; c += 256)
    atomicAdd(&pd[c], ps[c] + pe[c]);
  if (threadIdx.x == 0) atomicAdd(&cnt[d], 1.0f);
}

__global__ __launch_bounds__(256) void k_scatter_ee(
    const int* __restrict__ src, const int* __restrict__ dst,
    const float* __restrict__ he, float* agg, float* cnt, int ne) {
  int e = blockIdx.x;
  if (e >= ne) return;
  int s = src[e], d = dst[e];
  const float* ps = he + (size_t)s * HD;
  float* pd = agg + (size_t)d * HD;
  for (int c = threadIdx.x; c < HD; c += 256)
    atomicAdd(&pd[c], ps[c]);
  if (threadIdx.x == 0) atomicAdd(&cnt[d], 1.0f);
}

__global__ void k_norm(float* agg, const float* __restrict__ cnt, int nodes) {
  long long total = (long long)nodes * HD;
  for (long long idx = blockIdx.x * 256LL + threadIdx.x; idx < total;
       idx += (long long)gridDim.x * 256) {
    int i = (int)(idx >> 9);                 // HD == 512
    agg[idx] /= fmaxf(cnt[i], 1.0f);
  }
}

// ---------------- attention pooling kernels ----------------
__global__ __launch_bounds__(256) void k_rowdot(
    const float* __restrict__ q, const float* __restrict__ kmat,
    float* __restrict__ s, int M, float scale) {
  __shared__ float qs[HD];
  for (int i = threadIdx.x; i < HD; i += 256) qs[i] = q[i];
  __syncthreads();
  int wave = threadIdx.x >> 5, lane = threadIdx.x & 31;
  for (int row = blockIdx.x * 8 + wave; row < M; row += gridDim.x * 8) {
    const float* kr = kmat + (size_t)row * HD;
    float a = 0.0f;
    for (int c = lane; c < HD; c += 32) a += qs[c] * kr[c];
    for (int o = 16; o > 0; o >>= 1) a += __shfl_xor(a, o, 32);
    if (lane == 0) s[row] = a * scale;
  }
}

__global__ __launch_bounds__(1024) void k_smax_reduce(
    const float* __restrict__ s, float* __restrict__ red, int M) {
  __shared__ float sm[1024];
  float m = -INFINITY;
  for (int i = threadIdx.x; i < M; i += 1024) m = fmaxf(m, s[i]);
  sm[threadIdx.x] = m; __syncthreads();
  for (int o = 512; o > 0; o >>= 1) {
    if (threadIdx.x < o) sm[threadIdx.x] = fmaxf(sm[threadIdx.x], sm[threadIdx.x + o]);
    __syncthreads();
  }
  float mx = sm[0]; __syncthreads();
  float su = 0.0f;
  for (int i = threadIdx.x; i < M; i += 1024) su += expf(s[i] - mx);
  sm[threadIdx.x] = su; __syncthreads();
  for (int o = 512; o > 0; o >>= 1) {
    if (threadIdx.x < o) sm[threadIdx.x] += sm[threadIdx.x + o];
    __syncthreads();
  }
  if (threadIdx.x == 0) { red[0] = mx; red[1] = sm[0]; }
}

__global__ __launch_bounds__(512) void k_attnv(
    const float* __restrict__ s, const float* __restrict__ red,
    const float* __restrict__ v, float* av, int M) {
  __shared__ float ws[128];
  int base = blockIdx.x * 128;
  float mx = red[0], inv = 1.0f / red[1];
  if (threadIdx.x < 128) {
    int i = base + threadIdx.x;
    ws[threadIdx.x] = (i < M) ? expf(s[i] - mx) * inv : 0.0f;
  }
  __syncthreads();
  int h = threadIdx.x;
  float a = 0.0f;
  for (int r = 0; r < 128; r++) {
    int i = base + r;
    if (i >= M) break;
    a += ws[r] * v[(size_t)i * HD + h];
  }
  atomicAdd(&av[h], a);
}

// out[n] = sum_k x[k]*W[k*N+n] (+bias) ; single block, N<=512
__global__ __launch_bounds__(512) void k_gemv(
    const float* __restrict__ x, const float* __restrict__ W,
    const float* __restrict__ bias, float* __restrict__ out, int K, int N) {
  __shared__ float xs[HD];
  for (int i = threadIdx.x; i < K; i += 512) xs[i] = x[i];
  __syncthreads();
  int n = threadIdx.x;
  if (n >= N) return;
  float a = bias ? bias[n] : 0.0f;
  for (int k = 0; k < K; k++) a += xs[k] * W[(size_t)k * N + n];
  out[n] = a;
}

// ---------------- host side ----------------
static inline void gemm(hipStream_t st, const float* A, const bf16* Abf, int lda,
                        const bf16* Bt, const float* Cin, const float* bias,
                        float* D, int ldd, int M, int N, int K,
                        float scale, int leaky) {
  dim3 g((N + BN - 1) / BN, (M + BM - 1) / BM);
  k_gemm<<<g, 256, 0, st>>>(A, Abf, lda, Bt, Cin, bias, D, ldd, M, N, K, scale, leaky);
}

extern "C" void kernel_launch(void* const* d_in, const int* in_sizes, int n_in,
                              void* d_out, int out_size, void* d_ws, size_t ws_size,
                              hipStream_t stream) {
  (void)in_sizes; (void)n_in; (void)out_size; (void)ws_size;

  const float* x_win      = (const float*)d_in[0];
  const float* x_edge     = (const float*)d_in[1];
  const float* W_pre_win  = (const float*)d_in[2];
  const float* W_post_win = (const float*)d_in[3];
  const float* W_pre_edge = (const float*)d_in[4];
  const float* W_post_edge= (const float*)d_in[5];
  const float* conv_Wl    = (const float*)d_in[6];
  const float* conv_Wr    = (const float*)d_in[7];
  const float* conv_Wg    = (const float*)d_in[8];
  const float* conv_b     = (const float*)d_in[9];
  const float* econv_Wl   = (const float*)d_in[10];
  const float* econv_Wr   = (const float*)d_in[11];
  const float* econv_b    = (const float*)d_in[12];
  const float* pool_lin1_W= (const float*)d_in[13];
  const float* pool_lin1_b= (const float*)d_in[14];
  const float* pool_S     = (const float*)d_in[15];
  const float* pool_Wq    = (const float*)d_in[16];
  const float* pool_Wk    = (const float*)d_in[17];
  const float* pool_Wv    = (const float*)d_in[18];
  const float* pool_lin2_W= (const float*)d_in[19];
  const float* pool_lin2_b= (const float*)d_in[20];
  const float* lin_W      = (const float*)d_in[21];
  const float* lin_b      = (const float*)d_in[22];
  const int* ei_near  = (const int*)d_in[23];
  const int* ei_close = (const int*)d_in[24];
  const int* ei_sim   = (const int*)d_in[25];
  const int* ij_near  = (const int*)d_in[26];
  const int* ij_close = (const int*)d_in[27];
  const int* ij_sim   = (const int*)d_in[28];
  const int* ee_idx   = (const int*)d_in[29];
  float* out = (float*)d_out;

  // ---- workspace bump allocator ----
  uint8_t* base = (uint8_t*)d_ws;
  size_t off = 0;
  auto alloc = [&](size_t bytes) -> void* {
    void* p = base + off;
    off += (bytes + 255) & ~(size_t)255;
    return p;
  };

  bf16* WpreWinT   = (bf16*)alloc((size_t)HH * 2);
  bf16* WpostWinT  = (bf16*)alloc((size_t)HH * 2);
  bf16* WpreEdgeT  = (bf16*)alloc((size_t)HD * ECHP * 2);
  bf16* WpostEdgeT = (bf16*)alloc((size_t)HH * 2);
  bf16* convWlT    = (bf16*)alloc((size_t)NLAY * 3 * HH * 2);
  bf16* convWrSumT = (bf16*)alloc((size_t)NLAY * HH * 2);
  bf16* econvWlT   = (bf16*)alloc((size_t)NLAY * HH * 2);
  bf16* econvWrT   = (bf16*)alloc((size_t)NLAY * HH * 2);
  bf16* pool1T     = (bf16*)alloc((size_t)NLAY * HH * 2);
  bf16* WkT        = (bf16*)alloc((size_t)NLAY * HH * 2);
  bf16* WvT        = (bf16*)alloc((size_t)NLAY * HH * 2);
  bf16* linWT      = (bf16*)alloc((size_t)OUTD * HD * 2);
  bf16* xe_bf      = (bf16*)alloc((size_t)NEN * ECHP * 2);   // padded bf16 x_edge
  float* convWgSum = (float*)alloc((size_t)NLAY * HH * 4);
  float* convBSum  = (float*)alloc((size_t)NLAY * HD * 4);
  float* h_w   = (float*)alloc((size_t)NWN * HD * 4);
  float* h_e   = (float*)alloc((size_t)NEN * HD * 4);
  float* bufT  = (float*)alloc((size_t)NEN * HD * 4);   // t0/t1/h1/agg_e
  float* bufK  = (float*)alloc((size_t)NWN * HD * 4);   // k, then per-relation agg
  float* bufV  = (float*)alloc((size_t)NEN * HD * 4);   // v, then h_e gemm temp
  float* accb  = (float*)alloc((size_t)NWN * HD * 4);
  float* cnt   = (float*)alloc((size_t)NEN * 4);
  float* sv    = (float*)alloc((size_t)NWN * 4);
  float* qv    = (float*)alloc((size_t)HD * 4);
  float* av    = (float*)alloc((size_t)HD * 4);
  float* gvec  = (float*)alloc((size_t)HD * 4);
  float* gbias = (float*)alloc((size_t)HD * 4);
  float* red   = (float*)alloc(2 * 4);

  const float inv_sqrt_h = 1.0f / sqrtf((float)HD);

  // ---- weight prep (bf16, transposed, pre-summed across relations) ----
  k_wconvT<<<2048, 256, 0, stream>>>(W_pre_win,  WpreWinT,  HD, HD, HD);
  k_wconvT<<<2048, 256, 0, stream>>>(W_post_win, WpostWinT, HD, HD, HD);
  k_wconvT<<<2048, 256, 0, stream>>>(W_pre_edge, WpreEdgeT, ECHD, HD, ECHP);
  k_wconvT<<<2048, 256, 0, stream>>>(W_post_edge,WpostEdgeT,HD, HD, HD);
  for (int l = 0; l < NLAY; l++) {
    for (int r = 0; r < 3; r++)
      k_wconvT<<<2048, 256, 0, stream>>>(conv_Wl + (size_t)(l * 3 + r) * HH,
                                         convWlT + (size_t)(l * 3 + r) * HH, HD, HD, HD);
    k_wsum3T<<<2048, 256, 0, stream>>>(conv_Wr + (size_t)l * 3 * HH,
                                       convWrSumT + (size_t)l * HH, HD);
    k_add3<<<2048, 256, 0, stream>>>(conv_Wg + (size_t)l * 3 * HH,
                                     convWgSum + (size_t)l * HH, HH);
    k_add3<<<8, 256, 0, stream>>>(conv_b + (size_t)l * 3 * HD,
                                  convBSum + (size_t)l * HD, HD);
    k_wconvT<<<2048, 256, 0, stream>>>(econv_Wl + (size_t)l * HH, econvWlT + (size_t)l * HH, HD, HD, HD);
    k_wconvT<<<2048, 256, 0, stream>>>(econv_Wr + (size_t)l * HH, econvWrT + (size_t)l * HH, HD, HD, HD);
    k_wconvT<<<2048, 256, 0, stream>>>(pool_lin1_W + (size_t)l * HH, pool1T + (size_t)l * HH, HD, HD, HD);
    k_wconvT<<<2048, 256, 0, stream>>>(pool_Wk + (size_t)l * HH, WkT + (size_t)l * HH, HD, HD, HD);
    k_wconvT<<<2048, 256, 0, stream>>>(pool_Wv + (size_t)l * HH, WvT + (size_t)l * HH, HD, HD, HD);
  }
  k_wconvT<<<2048, 256, 0, stream>>>(lin_W, linWT, HD, OUTD, HD);
  k_convpad<<<8192, 256, 0, stream>>>(x_edge, xe_bf, NEN, ECHD, ECHP);

  // ---- pretransforms ----
  gemm(stream, x_win, nullptr, HD, WpreWinT,  nullptr, nullptr, bufT, HD, NWN, HD, HD, 1.0f, 1);
  gemm(stream, bufT,  nullptr, HD, WpostWinT, nullptr, nullptr, h_w,  HD, NWN, HD, HD, 1.0f, 1);
  gemm(stream, nullptr, xe_bf, ECHP, WpreEdgeT, nullptr, nullptr, bufT, HD, NEN, HD, ECHP, 1.0f, 1);
  gemm(stream, bufT,  nullptr, HD, WpostEdgeT, nullptr, nullptr, h_e, HD, NEN, HD, HD, 1.0f, 1);

  const int* rel_src[3] = { ei_near, ei_close, ei_sim };
  const int* rel_ij [3] = { ij_near, ij_close, ij_sim };

  for (int l = 0; l < NLAY; l++) {
    // ---- GMPool_I: single-seed attention pooling ----
    gemm(stream, h_w, nullptr, HD, pool1T + (size_t)l * HH, nullptr,
         pool_lin1_b + (size_t)l * HD, bufT, HD, NWN, HD, HD, 1.0f, 0);   // h1
    k_gemv<<<1, 512, 0, stream>>>(pool_S + (size_t)l * HD,
                                  pool_Wq + (size_t)l * HH, nullptr, qv, HD, HD);
    gemm(stream, bufT, nullptr, HD, WkT + (size_t)l * HH, nullptr, nullptr, bufK, HD, NWN, HD, HD, 1.0f, 0);
    gemm(stream, bufT, nullptr, HD, WvT + (size_t)l * HH, nullptr, nullptr, bufV, HD, NWN, HD, HD, 1.0f, 0);
    k_rowdot<<<256, 256, 0, stream>>>(qv, bufK, sv, NWN, inv_sqrt_h);
    k_smax_reduce<<<1, 1024, 0, stream>>>(sv, red, NWN);
    k_zero<<<1, 256, 0, stream>>>(av, HD);
    k_attnv<<<(NWN + 127) / 128, 512, 0, stream>>>(sv, red, bufV, av, NWN);
    k_gemv<<<1, 512, 0, stream>>>(av, pool_lin2_W + (size_t)l * HH,
                                  pool_lin2_b + (size_t)l * HD, gvec, HD, HD);
    // gbias = gvec @ sum_r Wg  +  sum_r b
    k_gemv<<<1, 512, 0, stream>>>(gvec, convWgSum + (size_t)l * HH,
                                  convBSum + (size_t)l * HD, gbias, HD, HD);

    // ---- hetero conv: acc = h_w @ (sum Wr) + gbias ; then += agg_r @ Wl_r ----
    gemm(stream, h_w, nullptr, HD, convWrSumT + (size_t)l * HH, nullptr, gbias,
         accb, HD, NWN, HD, HD, 1.0f, 0);
    for (int r = 0; r < 3; r++) {
      k_zero<<<4096, 256, 0, stream>>>(bufK, (long long)NWN * HD);
      k_zero<<<128, 256, 0, stream>>>(cnt, NWN);
      k_scatter_win<<<ERN, 256, 0, stream>>>(rel_src[r], rel_src[r] + ERN, rel_ij[r],
                                             h_w, h_e, bufK, cnt, ERN);
      k_norm<<<4096, 256, 0, stream>>>(bufK, cnt, NWN);
      if (r < 2) {
        gemm(stream, bufK, nullptr, HD, convWlT + (size_t)(l * 3 + r) * HH,
             accb, nullptr, accb, HD, NWN, HD, HD, 1.0f, 0);
      } else {
        // last relation: fuse /NREL and LeakyReLU, write new h_w
        gemm(stream, bufK, nullptr, HD, convWlT + (size_t)(l * 3 + 2) * HH,
             accb, nullptr, h_w, HD, NWN, HD, HD, 1.0f / 3.0f, 1);
      }
    }

    // ---- edge-graph SAGEConv (mean) ----
    k_zero<<<4096, 256, 0, stream>>>(bufT, (long long)NEN * HD);
    k_zero<<<128, 256, 0, stream>>>(cnt, NEN);
    k_scatter_ee<<<EEN, 256, 0, stream>>>(ee_idx, ee_idx + EEN, h_e, bufT, cnt, EEN);
    k_norm<<<4096, 256, 0, stream>>>(bufT, cnt, NEN);
    gemm(stream, bufT, nullptr, HD, econvWlT + (size_t)l * HH, nullptr,
         econv_b + (size_t)l * HD, bufV, HD, NEN, HD, HD, 1.0f, 0);
    gemm(stream, h_e, nullptr, HD, econvWrT + (size_t)l * HH, bufV, nullptr,
         h_e, HD, NEN, HD, HD, 1.0f, 1);
  }

  // ---- output projection ----
  gemm(stream, h_w, nullptr, HD, linWT, nullptr, lin_b, out, OUTD, NWN, OUTD, HD, 1.0f, 0);
}
